// SAGE_84275848282669
// MI455X (gfx1250) — matrix operand order repaired and checked by reading, not compiled, not run on hardware
//
#include <hip/hip_runtime.h>

#define NNODES 50000
#define NEDGES 800000
#define INCH   256
#define HIDC   256
#define OUTC   64
#define KBIG   512      // fused K = [agg | h]
#define LDSK   520      // padded LDS row stride (breaks 1KB-stride bank conflicts)
#define DROP_KEEP  0.8f
#define DROP_SCALE 1.25f

typedef __attribute__((ext_vector_type(16))) __bf16 v16bf;
typedef __attribute__((ext_vector_type(8)))  float  v8f;

union BF16Vec { unsigned short u[16]; uint4 q[2]; v16bf v; };

__device__ __forceinline__ unsigned short f2bf(float f) {
  unsigned int u = __float_as_uint(f);
  u += 0x7FFFu + ((u >> 16) & 1u);           // round-to-nearest-even
  return (unsigned short)(u >> 16);
}

__device__ __forceinline__ float hash_u01(unsigned int s) {
  s ^= s >> 16; s *= 0x7feb352dU; s ^= s >> 15; s *= 0x846ca68bU; s ^= s >> 16;
  return (float)(s >> 8) * (1.0f / 16777216.0f);
}

// gfx1250 async copy: global -> LDS, 16 bytes per lane, tracked by ASYNCcnt
__device__ __forceinline__ void async_g2l_b128(unsigned int lds_off, const void* gaddr) {
  asm volatile("global_load_async_to_lds_b128 %0, %1, off"
               :: "v"(lds_off), "v"(gaddr) : "memory");
}
__device__ __forceinline__ void wait_async0() {
  asm volatile("s_wait_asynccnt 0" ::: "memory");
}

// Build WT[Fout][512] (bf16, transposed) = [Wl ; Wr] columns
__global__ __launch_bounds__(256) void sage_cast_w(
    const float* __restrict__ Wl, const float* __restrict__ Wr,
    unsigned short* __restrict__ WT, int Fout) {
  int i = blockIdx.x * blockDim.x + threadIdx.x;
  if (i >= Fout * KBIG) return;
  int n = i / KBIG, k = i - n * KBIG;
  float v = (k < 256) ? Wl[(size_t)k * Fout + n] : Wr[(size_t)(k - 256) * Fout + n];
  WT[(size_t)n * KBIG + k] = f2bf(v);
}

__global__ __launch_bounds__(256) void sage_deg(
    const int* __restrict__ dst, float* __restrict__ deg, int E) {
  int e = blockIdx.x * blockDim.x + threadIdx.x;
  if (e < E) atomicAdd(&deg[dst[e]], 1.0f);
}

__global__ __launch_bounds__(256) void sage_drop_in(
    const float* __restrict__ x, float* __restrict__ h, int total, unsigned int seed) {
  int i = blockIdx.x * blockDim.x + threadIdx.x;
  if (i >= total) return;
  float u = hash_u01((unsigned int)i * 2654435761u ^ seed);
  h[i] = (u < DROP_KEEP) ? x[i] * DROP_SCALE : 0.0f;
}

// one block per edge; 256 channels scatter-added (L2-resident atomics)
__global__ __launch_bounds__(256) void sage_scatter(
    const float* __restrict__ h, const int* __restrict__ src,
    const int* __restrict__ dst, float* __restrict__ msg) {
  int e = blockIdx.x;
  int c = threadIdx.x;
  int s = src[e], d = dst[e];
  atomicAdd(&msg[(size_t)d * HIDC + c], h[(size_t)s * HIDC + c]);
}

// Abig[node][0:256)   = bf16(msg/deg)   (mean aggregation)
// Abig[node][256:512) = bf16(h)
__global__ __launch_bounds__(256) void sage_agg(
    const float* __restrict__ msg, const float* __restrict__ deg,
    const float* __restrict__ h, unsigned short* __restrict__ Abig, int total) {
  int i = blockIdx.x * blockDim.x + threadIdx.x;
  if (i >= total) return;
  int node = i >> 8, c = i & 255;
  float dg = fmaxf(deg[node], 1.0f);
  Abig[(size_t)node * KBIG + c]        = f2bf(msg[i] / dg);
  Abig[(size_t)node * KBIG + 256 + c]  = f2bf(h[i]);
}

// C[M x Fout] = Abig(bf16)[M x 512] @ WT^T(bf16) + bias ; optional ReLU+dropout
// 8 waves/block; wave = one 16-row tile x 64 cols (4x v8f accumulators)
__global__ __launch_bounds__(256, 2) void sage_gemm(
    const unsigned short* __restrict__ Abig,
    const unsigned short* __restrict__ WT,     // [Fout][512] bf16
    const float* __restrict__ bias,
    float* __restrict__ outF,
    int numTiles, int Fout, int mode, unsigned int seed) {
  __shared__ unsigned short ldsB[64 * LDSK];
  const int t = threadIdx.x;
  const int col0 = blockIdx.y * 64;

  __builtin_prefetch(bias + col0, 0, 0);       // global_prefetch_b8 (bias row)

  // stage 64 output columns of WT (64 x 512 bf16 = 64KB) into LDS via the
  // gfx1250 async copy engine (no VGPR round-trip), tracked by ASYNCcnt
  {
    const int n = t >> 2, q = t & 3;
    const unsigned short* gsrc = WT + (size_t)(col0 + n) * KBIG + q * 128;
    const unsigned int lbase =
        (unsigned int)(unsigned long long)(const void*)(ldsB + n * LDSK + q * 128);
#pragma unroll
    for (int i = 0; i < 16; ++i)
      async_g2l_b128(lbase + i * 16, gsrc + i * 8);
  }
  wait_async0();
  __syncthreads();

  const int wave = t >> 5;
  const int lane = t & 31;
  const int tile = blockIdx.x * 8 + wave;
  if (tile >= numTiles) return;               // wave-uniform: EXEC stays all-1s
  const int half = lane >> 4, ln = lane & 15;
  const int row0 = tile * 16;

  v8f zero = {0.f, 0.f, 0.f, 0.f, 0.f, 0.f, 0.f, 0.f};
  v8f acc[4];
#pragma unroll
  for (int i = 0; i < 4; ++i) acc[i] = zero;

  // A fragment 16x32 bf16: lane(row=ln, half) holds K {half*8+0..7, 16+half*8+0..7}
  const unsigned short* arow = Abig + (size_t)(row0 + ln) * KBIG + half * 8;

  for (int kk = 0; kk < KBIG; kk += 32) {
    BF16Vec a;
    a.q[0] = *(const uint4*)(arow + kk);
    a.q[1] = *(const uint4*)(arow + kk + 16);
#pragma unroll
    for (int ct = 0; ct < 4; ++ct) {
      // B fragment 32x16: lane(col=ln, half) holds K {half*8+0..7, 16+half*8+0..7}
      const unsigned short* bp = ldsB + (size_t)(ct * 16 + ln) * LDSK + kk + half * 8;
      BF16Vec b;
      b.q[0] = *(const uint4*)(bp);
      b.q[1] = *(const uint4*)(bp + 16);
      acc[ct] = __builtin_amdgcn_wmma_f32_16x16x32_bf16(
          false, a.v, false, b.v, (short)0, acc[ct], false, false);
    }
  }

  // epilogue: C/D layout -> VGPR v, lane l: row = v + 8*half, col = ln
#pragma unroll
  for (int ct = 0; ct < 4; ++ct) {
    const int col = col0 + ct * 16 + ln;
    const float bv = bias[col];
#pragma unroll
    for (int v = 0; v < 8; ++v) {
      const int row = row0 + half * 8 + v;
      float val = acc[ct][v] + bv;
      if (mode == 0) {  // hidden layers: ReLU + dropout
        val = fmaxf(val, 0.0f);
        float u = hash_u01((unsigned int)(row * Fout + col) * 2654435761u ^ seed);
        val = (u < DROP_KEEP) ? val * DROP_SCALE : 0.0f;
      }
      outF[(size_t)row * Fout + col] = val;
    }
  }
}

__global__ __launch_bounds__(256) void sage_loss(
    const float* __restrict__ conv, const int* __restrict__ y,
    const int* __restrict__ mask, float* __restrict__ red, int n) {
  int i = blockIdx.x * blockDim.x + threadIdx.x;
  if (i >= n) return;
  const float* row = conv + (size_t)i * OUTC;
  float mx = -1e30f;
#pragma unroll 8
  for (int c = 0; c < OUTC; ++c) mx = fmaxf(mx, row[c]);
  float s = 0.0f;
#pragma unroll 8
  for (int c = 0; c < OUTC; ++c) s += __expf(row[c] - mx);
  float lse = mx + __logf(s);
  float nll = lse - row[y[i]];
  float m = mask[i] ? 1.0f : 0.0f;
  int chunk = i / (n / 4);
  atomicAdd(&red[chunk], nll * m);
  atomicAdd(&red[4 + chunk], m);
}

__global__ void sage_final(const float* __restrict__ red, float* __restrict__ out) {
  float s = 0.0f;
#pragma unroll
  for (int i = 0; i < 4; ++i) s += red[i] / fmaxf(red[4 + i], 1.0f);
  out[0] = s * 0.25f;
}

extern "C" void kernel_launch(void* const* d_in, const int* in_sizes, int n_in,
                              void* d_out, int out_size, void* d_ws, size_t ws_size,
                              hipStream_t stream) {
  const float* x   = (const float*)d_in[0];
  const int* src   = (const int*)d_in[1];
  const int* dst   = (const int*)d_in[2];
  const int* y     = (const int*)d_in[3];
  const int* mask  = (const int*)d_in[4];
  const float* Wl0 = (const float*)d_in[5];
  const float* bl0 = (const float*)d_in[6];
  const float* Wr0 = (const float*)d_in[7];
  const float* Wl1 = (const float*)d_in[8];
  const float* bl1 = (const float*)d_in[9];
  const float* Wr1 = (const float*)d_in[10];
  const float* Wl2 = (const float*)d_in[11];
  const float* bl2 = (const float*)d_in[12];
  const float* Wr2 = (const float*)d_in[13];
  float* out = (float*)d_out;
  (void)in_sizes; (void)n_in; (void)out_size;

  char* ws = (char*)d_ws;
  size_t off = 0;
  auto take = [&](size_t bytes) -> char* {
    char* p = ws + off;
    off = (off + bytes + 255) & ~(size_t)255;
    return p;
  };
  float* h             = (float*)take((size_t)NNODES * HIDC * 4);
  float* msg           = (float*)take((size_t)NNODES * HIDC * 4);
  float* deg           = (float*)take((size_t)NNODES * 4);
  unsigned short* Abig = (unsigned short*)take((size_t)NNODES * KBIG * 2);
  unsigned short* WT0  = (unsigned short*)take((size_t)HIDC * KBIG * 2);
  unsigned short* WT1  = (unsigned short*)take((size_t)HIDC * KBIG * 2);
  unsigned short* WT2  = (unsigned short*)take((size_t)OUTC * KBIG * 2);
  float* conv          = (float*)take((size_t)NNODES * OUTC * 4);
  float* red           = (float*)take(8 * 4);
  if (off > ws_size) return;  // not enough scratch; deterministic no-op

  hipMemsetAsync(deg, 0, (size_t)NNODES * 4, stream);
  hipMemsetAsync(red, 0, 32, stream);

  sage_cast_w<<<(HIDC * KBIG + 255) / 256, 256, 0, stream>>>(Wl0, Wr0, WT0, HIDC);
  sage_cast_w<<<(HIDC * KBIG + 255) / 256, 256, 0, stream>>>(Wl1, Wr1, WT1, HIDC);
  sage_cast_w<<<(OUTC * KBIG + 255) / 256, 256, 0, stream>>>(Wl2, Wr2, WT2, OUTC);
  sage_deg<<<(NEDGES + 255) / 256, 256, 0, stream>>>(dst, deg, NEDGES);
  sage_drop_in<<<(NNODES * HIDC + 255) / 256, 256, 0, stream>>>(x, h, NNODES * HIDC, 0x1234567u);

  const int numTiles = NNODES / 16;  // 3125, exact
  const unsigned short* WTs[3] = {WT0, WT1, WT2};
  const float* biases[3] = {bl0, bl1, bl2};
  for (int L = 0; L < 3; ++L) {
    hipMemsetAsync(msg, 0, (size_t)NNODES * HIDC * 4, stream);
    sage_scatter<<<NEDGES, 256, 0, stream>>>(h, src, dst, msg);
    sage_agg<<<(NNODES * HIDC + 255) / 256, 256, 0, stream>>>(msg, deg, h, Abig, NNODES * HIDC);
    const int Fout = (L == 2) ? OUTC : HIDC;
    const int mode = (L == 2) ? 1 : 0;
    float* o = (L == 2) ? conv : h;
    dim3 grid((numTiles + 7) / 8, Fout / 64);
    sage_gemm<<<grid, 256, 0, stream>>>(Abig, WTs[L], biases[L], o,
                                        numTiles, Fout, mode, 0xBEEF0000u + L);
  }

  sage_loss<<<(NNODES + 255) / 256, 256, 0, stream>>>(conv, y, mask, red, NNODES);
  sage_final<<<1, 1, 0, stream>>>(red, out);
}
